// AIMNet_12876311954153
// MI455X (gfx1250) — matrix operand order repaired and checked by reading, not compiled, not run on hardware
//
#include <hip/hip_runtime.h>
#include <hip/hip_bf16.h>

typedef __attribute__((ext_vector_type(16))) _Float16 v16h;
typedef __attribute__((ext_vector_type(8)))  _Float16 v8h;
typedef __attribute__((ext_vector_type(8)))  float    v8f;
typedef __attribute__((ext_vector_type(4)))  int      v4i;

#define N_ATOMS   30000
#define N_EDGES   400000
#define N_EDGES_A 200000
#define N_ANGLES  500000
#define AFV_DIM   16
#define MAX_D     512   // largest layer width (emb input/hidden)

#if defined(__gfx1250__) && __has_builtin(__builtin_amdgcn_global_load_async_to_lds_b128)
#define HAVE_ASYNC_LDS 1
typedef __attribute__((address_space(1))) v4i* gv4i_p;   // global int4*
typedef __attribute__((address_space(3))) v4i* lv4i_p;   // LDS int4*
#else
#define HAVE_ASYNC_LDS 0
#endif

__device__ __forceinline__ float swishf(float x) {
    return x / (1.0f + __expf(-x));
}

// ---------------- WMMA fragment loaders (CDNA5 16x16x32 f16 layouts) ----------

// A-matrix 16x32 f16 from an LDS row-major tile [16 x din].
// lane%16 = M row; lanes 0-15 cover K {0..7,16..23}, lanes 16-31 K {8..15,24..31}.
// Per lane this is two contiguous 8-half (16B) runs -> 2x ds_load_b128.
__device__ __forceinline__ v16h load_afrag(const _Float16* lin, int din, int k0,
                                           int mrow, int hi) {
    const _Float16* base = lin + mrow * din + k0 + hi * 8;
    v8h lo = *(const v8h*)(base);        // K = k0 + hi*8 + {0..7}
    v8h hh = *(const v8h*)(base + 16);   // K = k0 + 16 + hi*8 + {0..7}
    v16h a;
#pragma unroll
    for (int h = 0; h < 8; ++h) { a[h] = lo[h]; a[h + 8] = hh[h]; }
    return a;
}

// B-matrix 32x16 f16 from pre-transposed f16 weights WT[dout][din] (column-major
// view of W[din][dout]). lane%16 = N col; lanes 0-15 K {0..7,16..23}, lanes 16-31 K {8..15,24..31}.
// Two contiguous 8-half (16B) vector loads.
__device__ __forceinline__ v16h load_bfrag(const _Float16* __restrict__ WT, int din,
                                           int j, int k0, int mrow, int hi) {
    const _Float16* base = WT + (long)(j + mrow) * din + k0 + hi * 8;
    v8h lo = *(const v8h*)(base);
    v8h hh = *(const v8h*)(base + 16);
    v16h b;
#pragma unroll
    for (int h = 0; h < 8; ++h) { b[h] = lo[h]; b[h + 8] = hh[h]; }
    return b;
}

// One dense layer over a 16-row LDS tile: lout = act(lin @ W + b), f16 result in LDS.
__device__ __forceinline__ void run_layer_wave(const _Float16* lin, int din,
                                               const _Float16* __restrict__ WT,
                                               const float* __restrict__ bs,
                                               int dout, _Float16* lout,
                                               bool act, int lane) {
    const int mrow = lane & 15;
    const int hi   = lane >> 4;
    for (int j = 0; j < dout; j += 16) {
        float bias = bs[j + mrow];      // accumulator N index = lane%16
        v8f c;
#pragma unroll
        for (int r = 0; r < 8; ++r) c[r] = bias;
        for (int k0 = 0; k0 < din; k0 += 32) {
            if (k0 + 32 < din)
                __builtin_prefetch(WT + (long)(j + mrow) * din + k0 + 32 + hi * 16, 0, 1);
            v16h a  = load_afrag(lin, din, k0, mrow, hi);
            v16h bb = load_bfrag(WT, din, j, k0, mrow, hi);
            c = __builtin_amdgcn_wmma_f32_16x16x32_f16(false, a, false, bb,
                                                       (short)0, c, false, false);
        }
#pragma unroll
        for (int r = 0; r < 8; ++r) {
            int   M = r + hi * 8;
            float v = act ? swishf(c[r]) : c[r];
            lout[M * dout + j + mrow] = (_Float16)v;
        }
    }
}

// Fused 3-layer MLP: out[M,d3] = L3(swish(L2(swish(L1(in)))))  (last layer linear)
__global__ void __launch_bounds__(32)
mlp3_wmma_kernel(const float* __restrict__ in, int Mtot, int din,
                 const _Float16* __restrict__ W1, const float* __restrict__ b1, int d1,
                 const _Float16* __restrict__ W2, const float* __restrict__ b2, int d2,
                 const _Float16* __restrict__ W3, const float* __restrict__ b3, int d3,
                 float* __restrict__ out) {
#if HAVE_ASYNC_LDS
    __shared__ __align__(32) float bufIn[16 * MAX_D];   // raw f32 staging (async DMA)
#endif
    __shared__ __align__(32) _Float16 bufA[16 * MAX_D];
    __shared__ __align__(32) _Float16 bufB[16 * MAX_D];
    const int lane = threadIdx.x;
    const int row0 = blockIdx.x * 16;

#if HAVE_ASYNC_LDS
    // async-DMA the f32 tile straight into LDS (16B chunks), then convert to f16
    const int dq = din >> 2;                       // 4-float chunks per row
    for (int c = lane; c < 16 * dq; c += 32) {
        int m = c / dq, k4 = (c - m * dq) << 2;
        int row = row0 + m;
        float* dst = &bufIn[m * din + k4];
        if (row < Mtot) {
            const float* src = in + (long)row * din + k4;
            __builtin_amdgcn_global_load_async_to_lds_b128(
                (gv4i_p)(v4i*)src, (lv4i_p)(v4i*)dst, 0, 0);
        } else {
            dst[0] = 0.0f; dst[1] = 0.0f; dst[2] = 0.0f; dst[3] = 0.0f;
        }
    }
    asm volatile("s_wait_asynccnt 0" ::: "memory");
    __syncthreads();
    for (int idx = lane; idx < 16 * din; idx += 32)
        bufA[idx] = (_Float16)bufIn[idx];
#else
    for (int idx = lane; idx < 16 * din; idx += 32) {
        int m = idx / din, k = idx - m * din;
        int row = row0 + m;
        bufA[idx] = (_Float16)((row < Mtot) ? in[(long)row * din + k] : 0.0f);
    }
#endif
    __syncthreads();
    run_layer_wave(bufA, din, W1, b1, d1, bufB, true, lane);
    __syncthreads();
    run_layer_wave(bufB, d1, W2, b2, d2, bufA, true, lane);
    __syncthreads();

    // final linear layer -> global f32
    const int mrow = lane & 15;
    const int hi   = lane >> 4;
    for (int j = 0; j < d3; j += 16) {
        float bias = b3[j + mrow];
        v8f c;
#pragma unroll
        for (int r = 0; r < 8; ++r) c[r] = bias;
        for (int k0 = 0; k0 < d2; k0 += 32) {
            v16h a  = load_afrag(bufA, d2, k0, mrow, hi);
            v16h bb = load_bfrag(W3, d2, j, k0, mrow, hi);
            c = __builtin_amdgcn_wmma_f32_16x16x32_f16(false, a, false, bb,
                                                       (short)0, c, false, false);
        }
#pragma unroll
        for (int r = 0; r < 8; ++r) {
            int row = row0 + r + hi * 8;
            if (row < Mtot) out[(long)row * d3 + j + mrow] = c[r];
        }
    }
}

// ---------------- weight prep: f32 [din,dout] -> f16 transposed [dout,din] -----

__global__ void k_prep_w(const float* __restrict__ W, _Float16* __restrict__ WT,
                         int din, int dout) {
    long i = (long)blockIdx.x * blockDim.x + threadIdx.x;
    if (i >= (long)din * dout) return;
    int k = (int)(i / dout), n = (int)(i - (long)k * dout);
    WT[(long)n * din + k] = (_Float16)W[i];
}

// ---------------- elementwise / gather / scatter kernels ----------------------

__global__ void k_gather_afv(const int* __restrict__ species,
                             const float* __restrict__ table,
                             float* __restrict__ afv, int n) {
    int i = blockIdx.x * blockDim.x + threadIdx.x;
    if (i >= n * AFV_DIM) return;
    int a = i >> 4, j = i & 15;
    afv[i] = table[species[a] * AFV_DIM + j];
}

__global__ void k_radial(const float* __restrict__ dist,
                         const float* __restrict__ sw,
                         float* __restrict__ rt, int E) {
    int e = blockIdx.x * blockDim.x + threadIdx.x;
    if (e >= E) return;
    float d = dist[e], s = sw[e];
#pragma unroll
    for (int r = 0; r < 16; ++r) {
        float shift = 0.8f + (float)r * 0.275f;     // linspace(0.8,5.2,17)[:-1]
        float t = d - shift;
        rt[e * 16 + r] = 0.25f * __expf(-16.0f * t * t) * s;
    }
}

__global__ void k_angular(const float* __restrict__ angles,
                          const float* __restrict__ da,
                          const float* __restrict__ swa,
                          const int* __restrict__ asrc,
                          const int* __restrict__ adst,
                          float* __restrict__ at, int A) {
    int i = blockIdx.x * blockDim.x + threadIdx.x;
    if (i >= A) return;
    int s = asrc[i], d = adst[i];
    float ang = angles[i];
    float d12 = 0.5f * (da[s] + da[d]);
    float sw  = 2.0f * swa[s] * swa[d];
    const float PI = 3.14159265358979f;
    float f1[4], f2[4];
#pragma unroll
    for (int z = 0; z < 4; ++z) {
        float shiftZ = PI / 8.0f + (float)z * (PI / 4.0f);
        float b = 0.5f + 0.5f * __cosf(ang - shiftZ);
        float p = b * b;                              // b^2
        p = p * p; p = p * p; p = p * p; p = p * p;   // b^32 (ZETA=32)
        f1[z] = p;
    }
#pragma unroll
    for (int r = 0; r < 4; ++r) {
        float shiftA = 0.8f + (float)r * 0.675f;    // linspace(0.8,3.5,5)[:-1]
        float t = d12 - shiftA;
        f2[r] = __expf(-8.0f * t * t);
    }
#pragma unroll
    for (int r = 0; r < 4; ++r)
#pragma unroll
        for (int z = 0; z < 4; ++z)
            at[i * 16 + r * 4 + z] = f1[z] * f2[r] * sw;   // reshape [A_DIV,A_SEC]
}

__global__ void k_afv12(const float* __restrict__ afv,
                        const int* __restrict__ edge_dst_a,
                        const int* __restrict__ asrc,
                        const int* __restrict__ adst,
                        float* __restrict__ afv12, int A) {
    int i = blockIdx.x * blockDim.x + threadIdx.x;
    if (i >= A) return;
    int a1 = edge_dst_a[asrc[i]], a2 = edge_dst_a[adst[i]];
#pragma unroll
    for (int j = 0; j < 16; ++j) {
        float x = afv[a1 * 16 + j], y = afv[a2 * 16 + j];
        afv12[(long)i * 32 + j]      = x * y;
        afv12[(long)i * 32 + 16 + j] = x + y;
    }
}

__global__ void k_zero(float* __restrict__ p, long n) {
    long i = (long)blockIdx.x * blockDim.x + threadIdx.x;
    if (i < n) p[i] = 0.0f;
}

__global__ void k_scatter_radial(const float* __restrict__ rt,
                                 const float* __restrict__ afv,
                                 const int* __restrict__ esrc,
                                 const int* __restrict__ edst,
                                 float* __restrict__ Gri, int E) {
    long t = (long)blockIdx.x * blockDim.x + threadIdx.x;   // (e, j)
    if (t >= (long)E * 16) return;
    int e = (int)(t >> 4), j = (int)(t & 15);
    float a = afv[edst[e] * 16 + j];
    long  base = (long)esrc[e] * 256 + j * 16;
#pragma unroll
    for (int k = 0; k < 16; ++k)
        atomicAdd(&Gri[base + k], a * rt[e * 16 + k]);
}

__global__ void k_scatter_angular(const float* __restrict__ at,
                                  const float* __restrict__ afv_ang,
                                  const int* __restrict__ central,
                                  float* __restrict__ Gai, int A) {
    long t = (long)blockIdx.x * blockDim.x + threadIdx.x;   // (a, c)
    if (t >= (long)A * 16) return;
    int a = (int)(t >> 4), c = (int)(t & 15);
    float v = afv_ang[(long)a * 16 + c];
    long  base = (long)central[a] * 256 + c * 16;
#pragma unroll
    for (int d = 0; d < 16; ++d)
        atomicAdd(&Gai[base + d], v * at[(long)a * 16 + d]);
}

__global__ void k_concat(const float* __restrict__ Gri,
                         const float* __restrict__ Gai,
                         float* __restrict__ Gin, int N) {
    long i = (long)blockIdx.x * blockDim.x + threadIdx.x;
    if (i >= (long)N * 512) return;
    int row = (int)(i >> 9), k = (int)(i & 511);
    Gin[i] = (k < 256) ? Gri[(long)row * 256 + k] : Gai[(long)row * 256 + (k - 256)];
}

__global__ void k_add(float* __restrict__ afv, const float* __restrict__ dafv, long n) {
    long i = (long)blockIdx.x * blockDim.x + threadIdx.x;
    if (i < n) afv[i] += dafv[i];
}

__global__ void k_copy(const float* __restrict__ src, float* __restrict__ dst, long n) {
    long i = (long)blockIdx.x * blockDim.x + threadIdx.x;
    if (i < n) dst[i] = src[i];
}

// ---------------- orchestration ------------------------------------------------

extern "C" void kernel_launch(void* const* d_in, const int* in_sizes, int n_in,
                              void* d_out, int out_size, void* d_ws, size_t ws_size,
                              hipStream_t stream) {
    const int N = N_ATOMS, E = N_EDGES, A = N_ANGLES;

    const int*   species  = (const int*)d_in[0];
    const float* dist     = (const float*)d_in[1];
    const float* sw       = (const float*)d_in[2];
    const int*   esrc     = (const int*)d_in[3];
    const int*   edst     = (const int*)d_in[4];
    const int*   edst_a   = (const int*)d_in[5];
    const float* angles   = (const float*)d_in[6];
    const float* dist_a   = (const float*)d_in[7];
    const float* sw_a     = (const float*)d_in[8];
    const int*   central  = (const int*)d_in[9];
    const int*   asrc     = (const int*)d_in[10];
    const int*   adst     = (const int*)d_in[11];

    // params flattened in dict order: afv_table, comb[3], emb[3], afvn[3], inter
    const float* afv_table = (const float*)d_in[12];
    auto P = [&](int i) { return (const float*)d_in[i]; };
    const int COMB0 = 13, EMB0 = 31, AFVN0 = 49, INTER0 = 67;

    // workspace layout (f32 region, then 32B-aligned f16 weight region)
    float* ws      = (float*)d_ws;
    float* afv     = ws;                        // N*16
    float* dafv    = afv     + (long)N * 16;    // N*16
    float* rt      = dafv    + (long)N * 16;    // E*16
    float* at      = rt      + (long)E * 16;    // A*16
    float* afv12   = at      + (long)A * 16;    // A*32
    float* afv_ang = afv12   + (long)A * 32;    // A*16
    float* Gri     = afv_ang + (long)A * 16;    // N*256
    float* Gai     = Gri     + (long)N * 256;   // N*256
    float* Gin     = Gai     + (long)N * 256;   // N*512
    float* fi      = Gin     + (long)N * 512;   // N*256
    _Float16* wt   = (_Float16*)((((uintptr_t)(fi + (long)N * 256)) + 31) & ~(uintptr_t)31);

    float* mi_out  = (float*)d_out;                       // N*128
    float* afv_out = (float*)d_out + (long)N * 128;       // N*16

    const int TB = 256;
    auto blocks = [&](long n) { return (int)((n + TB - 1) / TB); };
    const int tiles_atoms  = (N + 15) / 16;   // 1875
    const int tiles_angles = (A + 15) / 16;   // 31250

    // pre-transpose + f16-convert every dense weight (weights ~4MB f16, L2-resident)
    long wt_off = 0;
    auto prep = [&](int widx, int din, int dout) -> const _Float16* {
        _Float16* dst = wt + wt_off;
        long n = (long)din * dout;
        k_prep_w<<<blocks(n), TB, 0, stream>>>(P(widx), dst, din, dout);
        wt_off += n;
        return dst;
    };
    const _Float16 *cW1[3], *cW2[3], *cW3[3], *eW1[3], *eW2[3], *eW3[3],
                   *aW1[3], *aW2[3], *aW3[3], *iW1, *iW2, *iW3;
    for (int l = 0; l < 3; ++l) {
        cW1[l] = prep(COMB0 + l * 6 + 0, 32, 256);
        cW2[l] = prep(COMB0 + l * 6 + 2, 256, 128);
        cW3[l] = prep(COMB0 + l * 6 + 4, 128, 16);
        eW1[l] = prep(EMB0  + l * 6 + 0, 512, 512);
        eW2[l] = prep(EMB0  + l * 6 + 2, 512, 256);
        eW3[l] = prep(EMB0  + l * 6 + 4, 256, 256);
        aW1[l] = prep(AFVN0 + l * 6 + 0, 256, 256);
        aW2[l] = prep(AFVN0 + l * 6 + 2, 256, 256);
        aW3[l] = prep(AFVN0 + l * 6 + 4, 256, 16);
    }
    iW1 = prep(INTER0 + 0, 256, 256);
    iW2 = prep(INTER0 + 2, 256, 256);
    iW3 = prep(INTER0 + 4, 256, 128);

    k_gather_afv<<<blocks((long)N * 16), TB, 0, stream>>>(species, afv_table, afv, N);
    k_radial<<<blocks(E), TB, 0, stream>>>(dist, sw, rt, E);
    k_angular<<<blocks(A), TB, 0, stream>>>(angles, dist_a, sw_a, asrc, adst, at, A);

    for (int l = 0; l < 3; ++l) {
        const int cb = COMB0 + l * 6, eb = EMB0 + l * 6, ab = AFVN0 + l * 6;

        // afv12 from layer-start afv, then fused comb MLP (32->256->128->16)
        k_afv12<<<blocks(A), TB, 0, stream>>>(afv, edst_a, asrc, adst, afv12, A);
        mlp3_wmma_kernel<<<tiles_angles, 32, 0, stream>>>(
            afv12, A, 32,
            cW1[l], P(cb + 1), 256,
            cW2[l], P(cb + 3), 128,
            cW3[l], P(cb + 5), 16,
            afv_ang);

        // segment sums into Gri/Gai
        k_zero<<<blocks((long)N * 256), TB, 0, stream>>>(Gri, (long)N * 256);
        k_zero<<<blocks((long)N * 256), TB, 0, stream>>>(Gai, (long)N * 256);
        k_scatter_radial<<<blocks((long)E * 16), TB, 0, stream>>>(rt, afv, esrc, edst, Gri, E);
        k_scatter_angular<<<blocks((long)A * 16), TB, 0, stream>>>(at, afv_ang, central, Gai, A);
        k_concat<<<blocks((long)N * 512), TB, 0, stream>>>(Gri, Gai, Gin, N);

        // emb MLP (512->512->256->256) -> fi
        mlp3_wmma_kernel<<<tiles_atoms, 32, 0, stream>>>(
            Gin, N, 512,
            eW1[l], P(eb + 1), 512,
            eW2[l], P(eb + 3), 256,
            eW3[l], P(eb + 5), 256,
            fi);

        // afvn MLP (256->256->256->16) -> dafv ; afv += dafv
        mlp3_wmma_kernel<<<tiles_atoms, 32, 0, stream>>>(
            fi, N, 256,
            aW1[l], P(ab + 1), 256,
            aW2[l], P(ab + 3), 256,
            aW3[l], P(ab + 5), 16,
            dafv);
        k_add<<<blocks((long)N * 16), TB, 0, stream>>>(afv, dafv, (long)N * 16);

        if (l == 2) {
            // inter MLP (256->256->256->128) -> mi
            mlp3_wmma_kernel<<<tiles_atoms, 32, 0, stream>>>(
                fi, N, 256,
                iW1, P(INTER0 + 1), 256,
                iW2, P(INTER0 + 3), 256,
                iW3, P(INTER0 + 5), 128,
                mi_out);
        }
    }

    k_copy<<<blocks((long)N * 16), TB, 0, stream>>>(afv, afv_out, (long)N * 16);
}
